// MultiHeadAttention_46935402610696
// MI455X (gfx1250) — compile-verified, matrix operand order checked by reading, not run on hardware
//
#include <hip/hip_runtime.h>
#include <hip/hip_bf16.h>
#include <stdint.h>

// ---------------- problem constants ----------------
#define BATCH 4
#define SEQ   2048
#define HEADS 16
#define HDIM  64
#define HID   1024          // HEADS*HDIM
#define MROWS (BATCH*SEQ)   // 8192

typedef __bf16 bf16_t;
typedef __attribute__((ext_vector_type(8)))  __bf16 bf16x8;
typedef __attribute__((ext_vector_type(16))) __bf16 bf16x16;
typedef __attribute__((ext_vector_type(8)))  float  floatx8;

// Build a 16x32 bf16 WMMA fragment from two 16B LDS chunks.
// Lanes 0-15 hold K=[0..8)+[16..24), lanes 16-31 hold K=[8..16)+[24..32).
__device__ __forceinline__ bf16x16 ld_frag(const bf16_t* p0, const bf16_t* p1) {
    bf16x8 lo = *(const bf16x8*)p0;
    bf16x8 hi = *(const bf16x8*)p1;
    bf16x16 r;
#pragma unroll
    for (int i = 0; i < 8; ++i) { r[i] = lo[i]; r[i + 8] = hi[i]; }
    return r;
}

__device__ __forceinline__ floatx8 wmma_bf16(bf16x16 a, bf16x16 b, floatx8 c) {
    return __builtin_amdgcn_wmma_f32_16x16x32_bf16(false, a, false, b, (short)0, c,
                                                   false, false);
}

// Async copy of 64 contiguous bytes global -> LDS (CDNA5 ASYNCcnt path).
// GVS addressing: mem = saddr(base) + vaddr(u32 byte offset) + inst_offset;
// LDS dest = lds(u32) + inst_offset. No VGPR data movement.
__device__ __forceinline__ void async_copy64B(unsigned lds, unsigned voff,
                                              unsigned long long base) {
    asm volatile(
        "global_load_async_to_lds_b128 %0, %1, %2\n\t"
        "global_load_async_to_lds_b128 %0, %1, %2 offset:16\n\t"
        "global_load_async_to_lds_b128 %0, %1, %2 offset:32\n\t"
        "global_load_async_to_lds_b128 %0, %1, %2 offset:48"
        :: "v"(lds), "v"(voff), "s"(base) : "memory");
}

__device__ __forceinline__ void wait_async0() {
    asm volatile("s_wait_asynccnt 0x0" ::: "memory");
}

__device__ __forceinline__ void wait_ds0() {
    asm volatile("s_wait_dscnt 0x0" ::: "memory");
}

__device__ __forceinline__ unsigned lds_addr_u32(const void* p) {
    return (unsigned)(uintptr_t)p;   // low 32 bits of flat shared addr = LDS offset
}

// ============================================================
// GEMM: C[M,N] = A[M,K] (fp32) @ W[N,K]^T (fp32), bf16 WMMA math,
// double-buffered LDS so global loads of tile k+1 overlap WMMA of tile k.
// MODE 0: RoPE epilogue, store bf16 to (b,h,n,d)          (Q and K)
// MODE 1: store bf16 transposed to (b,h,d,n)              (V)
// MODE 2: store fp32 row-major (M,N)                      (final proj)
// Block: 256 threads (8 waves), 128x128 tile, BK=32,
// wave grid 2x4, each wave 64x32 = 4x2 WMMA tiles.
// ============================================================
template<int MODE>
__global__ __launch_bounds__(256)
void gemm_wmma(const float* __restrict__ A, const float* __restrict__ W,
               void* __restrict__ Out, int Kdim) {
    __shared__ bf16_t As[2][128][40];   // [buf][row][k], 80B row stride (16B aligned)
    __shared__ bf16_t Bs[2][128][40];

    const int tid  = threadIdx.x;
    const int bm   = blockIdx.y;
    const int bn   = blockIdx.x;
    const int w    = tid >> 5;
    const int lane = tid & 31;
    const int hi16 = lane >> 4;
    const int lr   = lane & 15;
    const int wm   = w >> 2;          // 0..1
    const int wn   = w & 3;           // 0..3

    floatx8 acc[4][2];
#pragma unroll
    for (int mt = 0; mt < 4; ++mt)
#pragma unroll
        for (int nt = 0; nt < 2; ++nt)
#pragma unroll
            for (int i = 0; i < 8; ++i) acc[mt][nt][i] = 0.0f;

    const int arow = tid >> 1;
    const int acb  = (tid & 1) * 16;
    const int c0   = hi16 * 8;
    const int c1   = 16 + hi16 * 8;

    const float* aBase = A + (size_t)(bm * 128 + arow) * Kdim + acb;
    const float* wBase = W + (size_t)(bn * 128 + arow) * Kdim + acb;

    float4 fa[4], fw[4];
    auto load_tile = [&](int k0) {
        const float4* ap = (const float4*)(aBase + k0);
        const float4* wp = (const float4*)(wBase + k0);
#pragma unroll
        for (int i = 0; i < 4; ++i) { fa[i] = ap[i]; fw[i] = wp[i]; }
    };
    auto store_tile = [&](int buf) {
        bf16x8 pa0, pa1, pw0, pw1;
        pa0[0]=(__bf16)fa[0].x; pa0[1]=(__bf16)fa[0].y; pa0[2]=(__bf16)fa[0].z; pa0[3]=(__bf16)fa[0].w;
        pa0[4]=(__bf16)fa[1].x; pa0[5]=(__bf16)fa[1].y; pa0[6]=(__bf16)fa[1].z; pa0[7]=(__bf16)fa[1].w;
        pa1[0]=(__bf16)fa[2].x; pa1[1]=(__bf16)fa[2].y; pa1[2]=(__bf16)fa[2].z; pa1[3]=(__bf16)fa[2].w;
        pa1[4]=(__bf16)fa[3].x; pa1[5]=(__bf16)fa[3].y; pa1[6]=(__bf16)fa[3].z; pa1[7]=(__bf16)fa[3].w;
        pw0[0]=(__bf16)fw[0].x; pw0[1]=(__bf16)fw[0].y; pw0[2]=(__bf16)fw[0].z; pw0[3]=(__bf16)fw[0].w;
        pw0[4]=(__bf16)fw[1].x; pw0[5]=(__bf16)fw[1].y; pw0[6]=(__bf16)fw[1].z; pw0[7]=(__bf16)fw[1].w;
        pw1[0]=(__bf16)fw[2].x; pw1[1]=(__bf16)fw[2].y; pw1[2]=(__bf16)fw[2].z; pw1[3]=(__bf16)fw[2].w;
        pw1[4]=(__bf16)fw[3].x; pw1[5]=(__bf16)fw[3].y; pw1[6]=(__bf16)fw[3].z; pw1[7]=(__bf16)fw[3].w;
        *(bf16x8*)&As[buf][arow][acb]     = pa0;
        *(bf16x8*)&As[buf][arow][acb + 8] = pa1;
        *(bf16x8*)&Bs[buf][arow][acb]     = pw0;
        *(bf16x8*)&Bs[buf][arow][acb + 8] = pw1;
    };

    const int nIter = Kdim >> 5;
    load_tile(0);
    store_tile(0);

    for (int it = 0; it < nIter; ++it) {
        __syncthreads();
        const int cur = it & 1;
        if (it + 1 < nIter) load_tile((it + 1) << 5);
        if (it + 2 < nIter) {   // prime L2 two tiles ahead (global_prefetch_b8)
            __builtin_prefetch(aBase + ((it + 2) << 5), 0, 3);
            __builtin_prefetch(wBase + ((it + 2) << 5), 0, 3);
        }

        bf16x16 af[4];
#pragma unroll
        for (int mt = 0; mt < 4; ++mt) {
            const int r = wm * 64 + mt * 16 + lr;
            af[mt] = ld_frag(&As[cur][r][c0], &As[cur][r][c1]);
        }
        bf16x16 bfr[2];
#pragma unroll
        for (int nt = 0; nt < 2; ++nt) {
            const int r = wn * 32 + nt * 16 + lr;
            bfr[nt] = ld_frag(&Bs[cur][r][c0], &Bs[cur][r][c1]);
        }
#pragma unroll
        for (int mt = 0; mt < 4; ++mt)
#pragma unroll
            for (int nt = 0; nt < 2; ++nt)
                acc[mt][nt] = wmma_bf16(af[mt], bfr[nt], acc[mt][nt]);

        if (it + 1 < nIter) store_tile((it + 1) & 1);
    }

    // ---- epilogue ----
#pragma unroll
    for (int mt = 0; mt < 4; ++mt) {
#pragma unroll
        for (int nt = 0; nt < 2; ++nt) {
#pragma unroll
            for (int j = 0; j < 8; ++j) {
                const int m_local = hi16 * 8 + j;
                const int rowG = bm * 128 + wm * 64 + mt * 16 + m_local;
                const int colG = bn * 128 + wn * 32 + nt * 16 + lr;
                float v = acc[mt][nt][j];
                if (MODE == 2) {
                    ((float*)Out)[(size_t)rowG * HID + colG] = v;
                } else {
                    const int pos  = rowG & (SEQ - 1);
                    const int bidx = rowG >> 11;
                    const int hh   = colG >> 6;
                    const int dd   = colG & 63;
                    if (MODE == 0) {
                        // RoPE: interleaved pair sits in the adjacent lane
                        float partner = __shfl_xor(v, 1, 32);
                        float ex  = (float)(dd & ~1) * (1.0f / (float)HDIM);
                        float inv = __powf(10000.0f, -ex);
                        float sn, cs;
                        __sincosf((float)pos * inv, &sn, &cs);
                        float rot = (dd & 1) ? partner : -partner;
                        float r   = v * cs + rot * sn;
                        ((bf16_t*)Out)[((((size_t)bidx * HEADS + hh) * SEQ + pos) << 6) + dd] =
                            (__bf16)r;
                    } else { // MODE 1: V transposed (b,h,d,n)
                        ((bf16_t*)Out)[(((size_t)bidx * HEADS + hh) * HDIM + dd) * SEQ + pos] =
                            (__bf16)v;
                    }
                }
            }
        }
    }
}

// ============================================================
// Flash attention: block = 128 threads (4 waves), 64 queries/block.
// K/V chunks double-buffered in LDS and staged with
// global_load_async_to_lds_b128 (ASYNCcnt), prefetched one chunk ahead.
// ============================================================
__global__ __launch_bounds__(128)
void flash_attn(const bf16_t* __restrict__ Qb, const bf16_t* __restrict__ Kb,
                const bf16_t* __restrict__ Vt, const unsigned char* __restrict__ pad,
                float* __restrict__ AttnOut,
                const int* __restrict__ pCausal, const int* __restrict__ pStart) {
    __shared__ bf16_t Qs[64][72];
    __shared__ bf16_t Ks[2][64][72];
    __shared__ bf16_t Vs[2][64][72];      // (d, key) layout
    __shared__ bf16_t Ps[4][16][72];      // per-wave P tiles

    const int tid  = threadIdx.x;
    const int w    = tid >> 5;
    const int lane = tid & 31;
    const int hi16 = lane >> 4;
    const int lr   = lane & 15;
    const int bh   = blockIdx.y;
    const int b    = bh >> 4;
    const int hh   = bh & 15;
    const int q0   = blockIdx.x * 64;
    const int is_causal = *pCausal;
    const int start_pos = *pStart;

    const unsigned long long baseQK =
        (unsigned long long)(Qb + (size_t)bh * SEQ * HDIM);          // same layout for K
    const unsigned long long baseK =
        (unsigned long long)(Kb + (size_t)bh * SEQ * HDIM);
    const unsigned long long baseV =
        (unsigned long long)(Vt + (size_t)bh * HDIM * SEQ);

    const int srow = tid >> 1;            // 0..63
    const int scb  = (tid & 1) * 32;      // element offset in row (32 bf16 = 64B)

    // ---- stage Q tile asynchronously ----
    async_copy64B(lds_addr_u32(&Qs[srow][scb]),
                  (unsigned)(((q0 + srow) * HDIM + scb) * sizeof(bf16_t)), baseQK);
    wait_async0();
    __syncthreads();

    const int c0 = hi16 * 8;
    bf16x16 qf[2];
    {
        const int qr = w * 16 + lr;
        qf[0] = ld_frag(&Qs[qr][c0],      &Qs[qr][c0 + 16]);
        qf[1] = ld_frag(&Qs[qr][32 + c0], &Qs[qr][32 + c0 + 16]);
    }

    floatx8 o[4];
#pragma unroll
    for (int ot = 0; ot < 4; ++ot)
#pragma unroll
        for (int i = 0; i < 8; ++i) o[ot][i] = 0.0f;
    float mrow[8], lrow[8];
#pragma unroll
    for (int j = 0; j < 8; ++j) { mrow[j] = -3.0e38f; lrow[j] = 0.0f; }

    int kend = SEQ;
    if (is_causal) {
        kend = q0 + 64;
        if (kend < start_pos) kend = start_pos;
        if (kend > SEQ) kend = SEQ;
    }
    const int nChunks = (kend + 63) >> 6;

    auto stage_chunk = [&](int k0, int buf) {
        async_copy64B(lds_addr_u32(&Ks[buf][srow][scb]),
                      (unsigned)(((k0 + srow) * HDIM + scb) * sizeof(bf16_t)), baseK);
        async_copy64B(lds_addr_u32(&Vs[buf][srow][scb]),
                      (unsigned)((srow * SEQ + k0 + scb) * sizeof(bf16_t)), baseV);
    };
    stage_chunk(0, 0);

    for (int c = 0; c < nChunks; ++c) {
        wait_async0();          // this wave's async copies for chunk c are done
        __syncthreads();        // everyone's are done
        if (c + 1 < nChunks) stage_chunk((c + 1) << 6, (c + 1) & 1);

        const int k0  = c << 6;
        const int buf = c & 1;

        // ---- S = Q @ K^T (4 tiles of 16 keys) ----
        floatx8 S[4];
#pragma unroll
        for (int nt = 0; nt < 4; ++nt) {
#pragma unroll
            for (int i = 0; i < 8; ++i) S[nt][i] = 0.0f;
            const int kr = nt * 16 + lr;
#pragma unroll
            for (int ks = 0; ks < 2; ++ks) {
                bf16x16 bk = ld_frag(&Ks[buf][kr][ks * 32 + c0],
                                     &Ks[buf][kr][ks * 32 + c0 + 16]);
                S[nt] = wmma_bf16(qf[ks], bk, S[nt]);
            }
        }

        // ---- scale + mask ----
        const float scale = 0.125f;   // 1/sqrt(64)
#pragma unroll
        for (int nt = 0; nt < 4; ++nt) {
            const int key = k0 + nt * 16 + lr;
            const bool pm = pad[(size_t)b * SEQ + key] != 0;
#pragma unroll
            for (int j = 0; j < 8; ++j) {
                const int qg = q0 + w * 16 + hi16 * 8 + j;
                const bool allow =
                    (!is_causal) || (key < start_pos) ||
                    ((qg >= start_pos) && (qg >= key));
                float sv = S[nt][j] * scale;
                S[nt][j] = (pm && allow) ? sv : -3.0e38f;
            }
        }

        // ---- online softmax (rows per half-wave, reduce over 16 lanes) ----
#pragma unroll
        for (int j = 0; j < 8; ++j) {
            float cm = fmaxf(fmaxf(S[0][j], S[1][j]), fmaxf(S[2][j], S[3][j]));
#pragma unroll
            for (int off = 1; off <= 8; off <<= 1)
                cm = fmaxf(cm, __shfl_xor(cm, off, 32));
            const float mn    = fmaxf(mrow[j], cm);
            const float alpha = __expf(mrow[j] - mn);
            mrow[j] = mn;
            float rs = 0.0f;
#pragma unroll
            for (int nt = 0; nt < 4; ++nt) {
                float p = __expf(S[nt][j] - mn);
                S[nt][j] = p;
                rs += p;
            }
#pragma unroll
            for (int off = 1; off <= 8; off <<= 1)
                rs += __shfl_xor(rs, off, 32);
            lrow[j] = lrow[j] * alpha + rs;
#pragma unroll
            for (int ot = 0; ot < 4; ++ot) o[ot][j] *= alpha;
        }

        // ---- restage P (C layout -> A layout) via wave-private LDS ----
#pragma unroll
        for (int nt = 0; nt < 4; ++nt)
#pragma unroll
            for (int j = 0; j < 8; ++j)
                Ps[w][hi16 * 8 + j][nt * 16 + lr] = (__bf16)S[nt][j];
        wait_ds0();   // wave-local: Ps is only read by this wave

        // ---- O += P @ V ----
#pragma unroll
        for (int ot = 0; ot < 4; ++ot) {
            const int vr = ot * 16 + lr;
#pragma unroll
            for (int ks = 0; ks < 2; ++ks) {
                bf16x16 ap = ld_frag(&Ps[w][lr][ks * 32 + c0],
                                     &Ps[w][lr][ks * 32 + c0 + 16]);
                bf16x16 bv = ld_frag(&Vs[buf][vr][ks * 32 + c0],
                                     &Vs[buf][vr][ks * 32 + c0 + 16]);
                o[ot] = wmma_bf16(ap, bv, o[ot]);
            }
        }
    }

    // ---- finalize: O / l -> attn buffer (b,n,hidden) fp32 ----
#pragma unroll
    for (int ot = 0; ot < 4; ++ot) {
#pragma unroll
        for (int j = 0; j < 8; ++j) {
            const int qg = q0 + w * 16 + hi16 * 8 + j;
            const float l = lrow[j];
            const float val = (l > 0.0f) ? (o[ot][j] / l) : 0.0f;
            AttnOut[((size_t)b * SEQ + qg) * HID + hh * HDIM + ot * 16 + lr] = val;
        }
    }
}

// ============================================================
extern "C" void kernel_launch(void* const* d_in, const int* in_sizes, int n_in,
                              void* d_out, int out_size, void* d_ws, size_t ws_size,
                              hipStream_t stream) {
    const float* x_q  = (const float*)d_in[0];
    const float* x_k  = (const float*)d_in[1];
    const float* x_v  = (const float*)d_in[2];
    const float* W_q  = (const float*)d_in[3];
    const float* W_k  = (const float*)d_in[4];
    const float* W_v  = (const float*)d_in[5];
    const float* W_o  = (const float*)d_in[6];
    const unsigned char* pad = (const unsigned char*)d_in[7];
    const int* pCausal = (const int*)d_in[8];
    const int* pStart  = (const int*)d_in[9];

    // workspace layout (bytes):
    //   [0,16M)   Q  bf16 (b,h,n,d)
    //   [16M,32M) K  bf16 (b,h,n,d)
    //   [32M,48M) Vt bf16 (b,h,d,n)
    //   [48M,80M) attn fp32 (b,n,hidden)
    char* ws = (char*)d_ws;
    const size_t SZ_BF = (size_t)MROWS * HID * sizeof(bf16_t);  // 16 MiB
    bf16_t* Qb   = (bf16_t*)(ws);
    bf16_t* Kb   = (bf16_t*)(ws + SZ_BF);
    bf16_t* Vtb  = (bf16_t*)(ws + 2 * SZ_BF);
    float*  attn = (float*)(ws + 3 * SZ_BF);

    dim3 gGrid(HID / 128, MROWS / 128);   // (8, 64)
    dim3 gBlock(256);
    gemm_wmma<0><<<gGrid, gBlock, 0, stream>>>(x_q, W_q, (void*)Qb,  HID);
    gemm_wmma<0><<<gGrid, gBlock, 0, stream>>>(x_k, W_k, (void*)Kb,  HID);
    gemm_wmma<1><<<gGrid, gBlock, 0, stream>>>(x_v, W_v, (void*)Vtb, HID);

    dim3 aGrid(SEQ / 64, BATCH * HEADS);  // (32, 64)
    flash_attn<<<aGrid, dim3(128), 0, stream>>>(Qb, Kb, Vtb, pad, attn, pCausal, pStart);

    gemm_wmma<2><<<gGrid, gBlock, 0, stream>>>(attn, W_o, d_out, HID);
}
